// HighLevelGuidedFrequencyFusion_5119601017231
// MI455X (gfx1250) — compile-verified, matrix operand order checked by reading, not run on hardware
//
#include <hip/hip_runtime.h>
#include <hip/hip_bf16.h>

typedef __attribute__((ext_vector_type(16))) __bf16 v16bf;
typedef __attribute__((ext_vector_type(8)))  float  v8f;

#define BZ 4
#define CC 64
#define HHD 256
#define WWD 256
#define NT 4096
#define TOK 1024
#define HIDN 128
#define NHEAD 4
#define DHD 32
#define PRI_D 64
#define KSEL 1024

__device__ __forceinline__ unsigned short f2bf(float f) {
  union { float f; unsigned u; } x; x.f = f;
  unsigned u = x.u;
  u += 0x7FFFu + ((u >> 16) & 1u);
  return (unsigned short)(u >> 16);
}
__device__ __forceinline__ float gelu_t(float x) {
  float x3 = x * x * x;
  return 0.5f * x * (1.f + tanhf(0.7978845608028654f * (x + 0.044715f * x3)));
}
__device__ __forceinline__ float sigm(float x) { return 1.f / (1.f + expf(-x)); }

// ---------------------------------------------------------------- pooling
__global__ __launch_bounds__(256) void mean_pool_kernel(const float* vis, const float* ir, float* pooled) {
  int bc = blockIdx.x;                 // b*128 + c2
  int b = bc >> 7, c2 = bc & 127;
  const float* src = (c2 < CC) ? (vis + ((size_t)(b * CC + c2)) * HHD * WWD)
                               : (ir  + ((size_t)(b * CC + (c2 - CC))) * HHD * WWD);
  float s = 0.f;
  for (int i = threadIdx.x; i < HHD * WWD; i += 256) s += src[i];
  __shared__ float red[256];
  red[threadIdx.x] = s; __syncthreads();
  for (int st = 128; st > 0; st >>= 1) {
    if (threadIdx.x < st) red[threadIdx.x] += red[threadIdx.x + st];
    __syncthreads();
  }
  if (threadIdx.x == 0) pooled[bc] = red[0] * (1.f / (HHD * WWD));
}

// ---------------------------------------------------------------- router + affine head
__global__ __launch_bounds__(512) void router_kernel(const float* pooled, const float* Wr, const float* br,
    const float* bank, const float* Wa1, const float* ba1, const float* Wa2, const float* ba2,
    float* intent, float* gamma, float* beta) {
  __shared__ float logit[BZ][4], pw[BZ][4], s_int[BZ][PRI_D], t1[BZ][HIDN];
  int tid = threadIdx.x;
  if (tid < BZ * 4) {
    int b = tid >> 2, p = tid & 3;
    float s = br[p];
    for (int c = 0; c < 2 * CC; ++c) s += pooled[b * 2 * CC + c] * Wr[c * 4 + p];
    logit[b][p] = s;
  }
  __syncthreads();
  if (tid < BZ) {
    int b = tid;
    float mx = logit[b][0];
    for (int p = 1; p < 4; ++p) mx = fmaxf(mx, logit[b][p]);
    float e[4], sm = 0.f;
    for (int p = 0; p < 4; ++p) { e[p] = expf(logit[b][p] - mx); sm += e[p]; }
    for (int p = 0; p < 4; ++p) pw[b][p] = e[p] / sm;
  }
  __syncthreads();
  if (tid < BZ * PRI_D) {
    int b = tid / PRI_D, r = tid % PRI_D;
    float s = 0.f;
    for (int p = 0; p < 4; ++p) s += pw[b][p] * bank[p * PRI_D + r];
    s_int[b][r] = s;
    intent[b * PRI_D + r] = s;
  }
  __syncthreads();
  {
    int b = tid / HIDN, h = tid % HIDN;
    float s = ba1[h];
    for (int r = 0; r < PRI_D; ++r) s += s_int[b][r] * Wa1[r * HIDN + h];
    t1[b][h] = gelu_t(s);
  }
  __syncthreads();
  {
    int b = tid / HIDN, o = tid % HIDN;
    float s = ba2[o];
    for (int h = 0; h < HIDN; ++h) s += t1[b][h] * Wa2[h * HIDN + o];
    if (o < CC) gamma[b * CC + o] = s; else beta[b * CC + (o - CC)] = s;
  }
}

__global__ __launch_bounds__(512) void intent_proj_kernel(const float* intent, const float* Wpri,
                                                          const float* Wint, float* PRIv, float* INTv) {
  int tid = threadIdx.x;
  int b = tid / HIDN, f = tid % HIDN;
  float s1 = 0.f, s2 = 0.f;
  for (int r = 0; r < PRI_D; ++r) {
    float iv = intent[b * PRI_D + r];
    s1 += iv * Wpri[r * HIDN + f];
    s2 += iv * Wint[r * HIDN + f];
  }
  PRIv[b * HIDN + f] = s1;
  INTv[b * HIDN + f] = s2;
}

// ---------------------------------------------------------------- 256-pt FFT lines (rows or cols)
__global__ __launch_bounds__(256) void fft256_kernel(const float* __restrict__ sre, const float* __restrict__ sim,
    float* __restrict__ dre, float* __restrict__ dim_,
    int transpose, int conj_in, int outmode, float scale) {
  __shared__ float lre[4][256], lim[4][256];
  int slot = threadIdx.x >> 6;
  int rl   = threadIdx.x & 63;
  int id   = blockIdx.x * 4 + slot;
  int img  = id >> 8, line = id & 255;
  size_t base   = (size_t)img * 65536 + (transpose ? (size_t)line : (size_t)line * 256);
  size_t stride = transpose ? 256 : 1;
  for (int t = rl; t < 256; t += 64) {
    int j = (int)(__brev((unsigned)t) >> 24);
    float re = sre[base + (size_t)t * stride];
    float im = sim ? sim[base + (size_t)t * stride] : 0.f;
    if (conj_in) im = -im;
    lre[slot][j] = re; lim[slot][j] = im;
  }
  __syncthreads();
  for (int len = 2; len <= 256; len <<= 1) {
    int half = len >> 1;
    for (int t = rl; t < 128; t += 64) {
      int blk = t / half, j = t % half;
      int pos = blk * len + j;
      float ang = -6.283185307179586f * (float)j / (float)len;
      float wr = cosf(ang), wi = sinf(ang);
      float ar = lre[slot][pos],        ai = lim[slot][pos];
      float br_ = lre[slot][pos + half], bi = lim[slot][pos + half];
      float tr = br_ * wr - bi * wi, ti = br_ * wi + bi * wr;
      lre[slot][pos] = ar + tr;         lim[slot][pos] = ai + ti;
      lre[slot][pos + half] = ar - tr;  lim[slot][pos + half] = ai - ti;
    }
    __syncthreads();
  }
  for (int t = rl; t < 256; t += 64) {
    float re = lre[slot][t], im = lim[slot][t];
    size_t o = base + (size_t)t * stride;
    if (outmode == 0)      { dre[o] = re; dim_[o] = im; }
    else if (outmode == 1) { dre[o] = sqrtf(re * re + im * im); dim_[o] = atan2f(im, re); }
    else                   { dre[o] = re * scale; }
  }
}

// ---------------------------------------------------------------- token packing (patchify -> bf16)
__global__ __launch_bounds__(256) void pack_tokens_kernel(const float* srcV, const float* srcI, unsigned short* cat) {
  size_t i = (size_t)blockIdx.x * 256 + threadIdx.x;   // (b*4096+t)*2048 + f2
  if (i >= (size_t)BZ * NT * 2048) return;
  int f2 = (int)(i & 2047);
  int t  = (int)((i >> 11) & 4095);
  int b  = (int)(i >> 23);
  int f  = f2 & 1023;
  const float* src = (f2 < 1024) ? srcV : srcI;
  int c = f >> 4, py = (f >> 2) & 3, px = f & 3;
  int gy = t >> 6, gx = t & 63;
  float v = src[(((size_t)(b * CC + c)) * HHD + gy * 4 + py) * WWD + gx * 4 + px];
  cat[i] = f2bf(v);
}

__global__ __launch_bounds__(256) void cvt_bf16_kernel(const float* src, unsigned short* dst, long n) {
  long i = (long)blockIdx.x * 256 + threadIdx.x;
  if (i < n) dst[i] = f2bf(src[i]);
}

// ---------------------------------------------------------------- WMMA fragment helpers
union FragU { v16bf v; unsigned u[8]; };

__device__ __forceinline__ void load_frag(FragU& f, const unsigned short* p, int kb8) {
#pragma unroll
  for (int v = 0; v < 8; ++v) {
    int k2 = ((v >= 4) ? 16 : 0) + kb8 + (v & 3) * 2;
    f.u[v] = (unsigned)p[k2] | ((unsigned)p[k2 + 1] << 16);
  }
}

// ---------------------------------------------------------------- FAST bf16 WMMA GEMM
// Requires: M%64==0, N%64==0, K%32==0, grid exact. No guards, LDS double
// buffer, one barrier per K-step, global loads for step s+1 in flight while
// step s computes.
__global__ __launch_bounds__(128) void gemm_bf16_fast_kernel(
    const unsigned short* __restrict__ A, const unsigned short* __restrict__ Bm, float* __restrict__ C,
    int K, int lda, int ldb, int ldc,
    long long sAb, long long sAh, long long sBb, long long sBh, long long sCb, long long sCh, int nh,
    float alpha, int accumulate) {
  __shared__ __align__(16) unsigned short As[2][64][40];
  __shared__ __align__(16) unsigned short Bt[2][64][40];   // Bt[buf][col][k]
  int z = blockIdx.z, bb = z / nh, hh = z % nh;
  A  += bb * sAb + hh * sAh;
  Bm += bb * sBb + hh * sBh;
  C  += bb * sCb + hh * sCh;
  int m0 = blockIdx.y * 64, n0 = blockIdx.x * 64;
  int tid = threadIdx.x, lane = tid & 31, wave = tid >> 5;
  int wm = (wave >> 1) * 32, wn = (wave & 1) * 32;

  int arow = tid >> 1, aseg = tid & 1;
  int bkr  = tid >> 2, bseg = tid & 3;
  const unsigned short* aptr = A + (size_t)(m0 + arow) * lda + aseg * 16;
  const unsigned short* bptr = Bm + (size_t)bkr * ldb + n0 + bseg * 16;
  size_t bstep = (size_t)32 * ldb;

  v8f acc[2][2];
#pragma unroll
  for (int a2 = 0; a2 < 2; ++a2)
#pragma unroll
    for (int b2 = 0; b2 < 2; ++b2)
#pragma unroll
      for (int e = 0; e < 8; ++e) acc[a2][b2][e] = 0.f;

  // prologue: tile 0 into registers
  uint4 ra0 = *(const uint4*)aptr, ra1 = *(const uint4*)(aptr + 8);
  uint4 rb0 = *(const uint4*)bptr, rb1 = *(const uint4*)(bptr + 8);
  aptr += 32; bptr += bstep;

  int nsteps = K >> 5;
  int cur = 0;
  for (int s = 0; s < nsteps; ++s) {
    // store staged tile into buf[cur]
    unsigned short* adst = &As[cur][arow][aseg * 16];
    *(uint4*)adst       = ra0;
    *(uint4*)(adst + 8) = ra1;
    {
      unsigned short tmp[16];
      *(uint4*)tmp       = rb0;
      *(uint4*)(tmp + 8) = rb1;
#pragma unroll
      for (int e = 0; e < 16; ++e) Bt[cur][bseg * 16 + e][bkr] = tmp[e];
    }
    if (s + 1 < nsteps) {  // next tile into registers (overlaps with compute below)
      ra0 = *(const uint4*)aptr; ra1 = *(const uint4*)(aptr + 8);
      rb0 = *(const uint4*)bptr; rb1 = *(const uint4*)(bptr + 8);
      aptr += 32; bptr += bstep;
      __builtin_prefetch(aptr, 0, 1);  // distance-2 A prefetch -> global_prefetch_b8
    }
    __syncthreads();
    FragU fa[2], fb[2];
    int kb8 = ((lane >> 4) & 1) * 8;
    load_frag(fa[0], &As[cur][wm      + (lane & 15)][0], kb8);
    load_frag(fa[1], &As[cur][wm + 16 + (lane & 15)][0], kb8);
    load_frag(fb[0], &Bt[cur][wn      + (lane & 15)][0], kb8);
    load_frag(fb[1], &Bt[cur][wn + 16 + (lane & 15)][0], kb8);
#pragma unroll
    for (int mi = 0; mi < 2; ++mi)
#pragma unroll
      for (int ni = 0; ni < 2; ++ni)
        acc[mi][ni] = __builtin_amdgcn_wmma_f32_16x16x32_bf16(
            false, fa[mi].v, false, fb[ni].v, (short)0, acc[mi][ni], false, false);
    cur ^= 1;
  }
#pragma unroll
  for (int mi = 0; mi < 2; ++mi)
#pragma unroll
    for (int ni = 0; ni < 2; ++ni)
#pragma unroll
      for (int r = 0; r < 8; ++r) {
        int mr = m0 + wm + mi * 16 + ((lane >> 4) * 8 + r);
        int nc = n0 + wn + ni * 16 + (lane & 15);
        float v = alpha * acc[mi][ni][r];
        size_t o = (size_t)mr * ldc + nc;
        C[o] = accumulate ? (C[o] + v) : v;
      }
}

// ---------------------------------------------------------------- N=32 bf16 WMMA GEMM (attn @ V)
// Requires: M%64==0, N==32 exactly, K%32==0, grid (1, M/64, batches). Four
// waves each own a 16x32 row band: 1 A-frag, 2 B-frags, 2 WMMAs per K-step.
__global__ __launch_bounds__(128) void gemm_bf16_n32_kernel(
    const unsigned short* __restrict__ A, const unsigned short* __restrict__ Bm, float* __restrict__ C,
    int K, int lda, int ldb, int ldc,
    long long sAb, long long sAh, long long sBb, long long sBh, long long sCb, long long sCh, int nh,
    float alpha) {
  __shared__ __align__(16) unsigned short As[64][40];
  __shared__ __align__(16) unsigned short Bt[32][40];   // Bt[col][k]
  int z = blockIdx.z, bb = z / nh, hh = z % nh;
  A  += bb * sAb + hh * sAh;
  Bm += bb * sBb + hh * sBh;
  C  += bb * sCb + hh * sCh;
  int m0 = blockIdx.y * 64;
  int tid = threadIdx.x, lane = tid & 31, wave = tid >> 5;
  int wm = wave * 16;

  int arow = tid >> 1, aseg = tid & 1;
  int bkr  = tid >> 2, bseg = tid & 3;      // k-row 0..31, 8-col segment
  unsigned short* adst = &As[arow][aseg * 16];

  v8f acc[2];
#pragma unroll
  for (int b2 = 0; b2 < 2; ++b2)
#pragma unroll
    for (int e = 0; e < 8; ++e) acc[b2][e] = 0.f;

  for (int kt = 0; kt < K; kt += 32) {
    {
      const unsigned short* gp = A + (size_t)(m0 + arow) * lda + kt + aseg * 16;
      *(uint4*)adst       = *(const uint4*)gp;
      *(uint4*)(adst + 8) = *(const uint4*)(gp + 8);
    }
    {
      const unsigned short* gp = Bm + (size_t)(kt + bkr) * ldb + bseg * 8;
      uint4 q = *(const uint4*)gp;
      unsigned short tmp[8];
      *(uint4*)tmp = q;
#pragma unroll
      for (int e = 0; e < 8; ++e) Bt[bseg * 8 + e][bkr] = tmp[e];
    }
    __syncthreads();
    FragU fa, fb0, fb1;
    int kb8 = ((lane >> 4) & 1) * 8;
    load_frag(fa,  &As[wm + (lane & 15)][0], kb8);
    load_frag(fb0, &Bt[(lane & 15)][0],      kb8);
    load_frag(fb1, &Bt[16 + (lane & 15)][0], kb8);
    acc[0] = __builtin_amdgcn_wmma_f32_16x16x32_bf16(false, fa.v, false, fb0.v, (short)0, acc[0], false, false);
    acc[1] = __builtin_amdgcn_wmma_f32_16x16x32_bf16(false, fa.v, false, fb1.v, (short)0, acc[1], false, false);
    __syncthreads();
  }
#pragma unroll
  for (int ni = 0; ni < 2; ++ni)
#pragma unroll
    for (int r = 0; r < 8; ++r) {
      int mr = m0 + wm + ((lane >> 4) * 8 + r);
      int nc = ni * 16 + (lane & 15);
      C[(size_t)mr * ldc + nc] = alpha * acc[ni][r];
    }
}

// ---------------------------------------------------------------- token scoring
__global__ __launch_bounds__(128) void score_kernel(const float* Hbuf, const float* PRIv,
    const float* Wco, const float* bs, const float* Wout, const float* bout_, float* score) {
  int row = blockIdx.x;  // b*4096+t
  int b = row >> 12, t = row & 4095;
  int f = threadIdx.x;
  int gy = t >> 6, gx = t & 63;
  float cy = (gy + 0.5f) / 64.f, cx = (gx + 0.5f) / 64.f;
  float h = Hbuf[(size_t)row * HIDN + f] + PRIv[b * HIDN + f] + cy * Wco[f] + cx * Wco[HIDN + f] + bs[f];
  float pr = gelu_t(h) * Wout[f];
  __shared__ float red[128];
  red[f] = pr; __syncthreads();
  for (int st = 64; st > 0; st >>= 1) {
    if (f < st) red[f] += red[f + st];
    __syncthreads();
  }
  if (f == 0) score[row] = red[0] + bout_[0];
}

// ---------------------------------------------------------------- top-k via LDS bitonic sort (descending)
__global__ __launch_bounds__(1024) void topk_kernel(const float* score, int* idxout) {
  int b = blockIdx.x;
  __shared__ float v[4096];
  __shared__ int ix[4096];
  int tid = threadIdx.x;
  for (int t = tid; t < 4096; t += 1024) { v[t] = score[b * 4096 + t]; ix[t] = t; }
  __syncthreads();
  for (int k = 2; k <= 4096; k <<= 1) {
    for (int j = k >> 1; j > 0; j >>= 1) {
      for (int t = tid; t < 2048; t += 1024) {
        int i = ((t / j) * (j << 1)) + (t % j);
        int p = i + j;
        bool dir = ((i & k) == 0);
        float a = v[i], c = v[p];
        bool sw = dir ? (a < c) : (a > c);
        if (sw) { v[i] = c; v[p] = a; int ti = ix[i]; ix[i] = ix[p]; ix[p] = ti; }
      }
      __syncthreads();
    }
  }
  if (tid < KSEL) idxout[b * KSEL + tid] = ix[tid];
}

// ---------------------------------------------------------------- gather selected tokens
__global__ __launch_bounds__(256) void gather_kernel(const unsigned short* cat, const float* srcV, const int* idx,
                                                     unsigned short* vsb, unsigned short* irsb, float* fused) {
  size_t i = (size_t)blockIdx.x * 256 + threadIdx.x;  // (b*1024+j)*1024+f
  if (i >= (size_t)BZ * KSEL * TOK) return;
  int f = (int)(i & 1023);
  int j = (int)((i >> 10) & 1023);
  int b = (int)(i >> 20);
  int t = idx[b * KSEL + j];
  size_t crow = ((size_t)(b * NT + t)) * 2048;
  vsb[i]  = cat[crow + f];
  irsb[i] = cat[crow + 1024 + f];
  int c = f >> 4, py = (f >> 2) & 3, px = f & 3, gy = t >> 6, gx = t & 63;
  fused[i] = srcV[(((size_t)(b * CC + c)) * HHD + gy * 4 + py) * WWD + gx * 4 + px];
}

__global__ __launch_bounds__(256) void kT_kernel(const float* KF, unsigned short* KT) {
  size_t i = (size_t)blockIdx.x * 256 + threadIdx.x;  // ((b*4+h)*32+d)*1024+n
  if (i >= (size_t)BZ * NHEAD * DHD * KSEL) return;
  int n = (int)(i & 1023);
  int d = (int)((i >> 10) & 31);
  int h = (int)((i >> 15) & 3);
  int b = (int)(i >> 17);
  KT[i] = f2bf(KF[((size_t)(b * KSEL + n)) * HIDN + h * DHD + d]);
}

__global__ __launch_bounds__(256) void softmax_kernel(const float* S, unsigned short* A) {
  size_t row = blockIdx.x;
  const float* s = S + row * KSEL;
  unsigned short* a = A + row * KSEL;
  int tid = threadIdx.x;
  float lv[4];
  float mx = -3.4e38f;
  for (int q = 0; q < 4; ++q) { lv[q] = s[tid + q * 256]; mx = fmaxf(mx, lv[q]); }
  __shared__ float red[256];
  red[tid] = mx; __syncthreads();
  for (int st = 128; st > 0; st >>= 1) {
    if (tid < st) red[tid] = fmaxf(red[tid], red[tid + st]);
    __syncthreads();
  }
  mx = red[0]; __syncthreads();
  float sm = 0.f;
  for (int q = 0; q < 4; ++q) { lv[q] = expf(lv[q] - mx); sm += lv[q]; }
  red[tid] = sm; __syncthreads();
  for (int st = 128; st > 0; st >>= 1) {
    if (tid < st) red[tid] += red[tid + st];
    __syncthreads();
  }
  float inv = 1.f / red[0];
  for (int q = 0; q < 4; ++q) a[tid + q * 256] = f2bf(lv[q] * inv);
}

__global__ __launch_bounds__(256) void o_post_kernel(float* OF, const float* INTv, unsigned short* OB) {
  size_t i = (size_t)blockIdx.x * 256 + threadIdx.x;  // (b*1024+j)*128+f
  if (i >= (size_t)BZ * KSEL * HIDN) return;
  int f = (int)(i & 127);
  int b = (int)(i >> 17);
  float v = OF[i] + INTv[b * HIDN + f];
  OF[i] = v;
  OB[i] = f2bf(v);
}

// ---------------------------------------------------------------- bypass blend (unpatchify on the fly)
__global__ __launch_bounds__(256) void blend_kernel(const float* G, const float* bbyp,
                                                    const float* srcV, const float* srcI, float* target) {
  size_t i = (size_t)blockIdx.x * 256 + threadIdx.x;  // (b*4096+t)*1024+f
  if (i >= (size_t)BZ * NT * TOK) return;
  int f = (int)(i & 1023);
  int t = (int)((i >> 10) & 4095);
  int b = (int)(i >> 22);
  float g = sigm(G[i] + bbyp[f]);
  int c = f >> 4, py = (f >> 2) & 3, px = f & 3, gy = t >> 6, gx = t & 63;
  size_t pix = (((size_t)(b * CC + c)) * HHD + gy * 4 + py) * WWD + gx * 4 + px;
  target[pix] = g * srcV[pix] + (1.f - g) * srcI[pix];
}

__global__ __launch_bounds__(256) void scatter_kernel(const float* fused, const int* idx, float* target) {
  size_t i = (size_t)blockIdx.x * 256 + threadIdx.x;  // (b*1024+j)*1024+f
  if (i >= (size_t)BZ * KSEL * TOK) return;
  int f = (int)(i & 1023);
  int j = (int)((i >> 10) & 1023);
  int b = (int)(i >> 20);
  int t = idx[b * KSEL + j];
  int c = f >> 4, py = (f >> 2) & 3, px = f & 3, gy = t >> 6, gx = t & 63;
  size_t pix = (((size_t)(b * CC + c)) * HHD + gy * 4 + py) * WWD + gx * 4 + px;
  target[pix] = fused[i];
}

__global__ __launch_bounds__(256) void spec_kernel(const float* famp, const float* fph, float* re, float* im) {
  size_t i = (size_t)blockIdx.x * 256 + threadIdx.x;
  if (i >= (size_t)BZ * CC * HHD * WWD) return;
  float a = famp[i], p = fph[i];
  re[i] = a * cosf(p);
  im[i] = a * sinf(p);
}

// ---------------------------------------------------------------- 3x3 conv with LDS tile (+fused epilogue)
__global__ __launch_bounds__(256) void conv3_kernel(const float* __restrict__ x, const float* __restrict__ w,
    const float* __restrict__ bias, float* __restrict__ y, int mode,
    const float* gamma, const float* beta, const float* vis, const float* ir) {
  __shared__ float tile[18][18];
  int tid = threadIdx.x;
  int tx = tid & 15, ty = tid >> 4;
  int x0 = blockIdx.x * 16, y0 = blockIdx.y * 16;
  int bc = blockIdx.z;
  int co = bc & 63, b = bc >> 6;
  float acc = 0.f;
  for (int ci = 0; ci < CC; ++ci) {
    const float* xp = x + ((size_t)(b * CC + ci)) * HHD * WWD;
    for (int l = tid; l < 18 * 18; l += 256) {
      int ly = l / 18, lx = l % 18;
      int gy = y0 + ly - 1, gx = x0 + lx - 1;
      tile[ly][lx] = (gy >= 0 && gy < HHD && gx >= 0 && gx < WWD) ? xp[(size_t)gy * WWD + gx] : 0.f;
    }
    __syncthreads();
    const float* wp = w + ((size_t)(co * CC + ci)) * 9;
#pragma unroll
    for (int ky = 0; ky < 3; ++ky)
#pragma unroll
      for (int kx = 0; kx < 3; ++kx)
        acc += tile[ty + ky][tx + kx] * wp[ky * 3 + kx];
    __syncthreads();
  }
  acc += bias[co];
  size_t o = (((size_t)(b * CC + co)) * HHD + y0 + ty) * WWD + x0 + tx;
  if (mode == 0) {
    y[o] = fmaxf(acc, 0.f);
  } else {
    float gm = gamma[b * CC + co], bt = beta[b * CC + co];
    y[o] = acc * (1.f + 0.1f * tanhf(gm)) + 0.1f * bt + 0.5f * (vis[o] + ir[o]);
  }
}

// ================================================================ host orchestration
extern "C" void kernel_launch(void* const* d_in, const int* in_sizes, int n_in,
                              void* d_out, int out_size, void* d_ws, size_t ws_size,
                              hipStream_t stream) {
  (void)in_sizes; (void)n_in; (void)out_size; (void)ws_size;
  const float* vis  = (const float*)d_in[0];
  const float* ir   = (const float*)d_in[1];
  const float* bank = (const float*)d_in[2];
  const float* Wr   = (const float*)d_in[3];
  const float* br   = (const float*)d_in[4];
  const float* Wtok = (const float*)d_in[5];
  const float* Wpri = (const float*)d_in[6];
  const float* Wco  = (const float*)d_in[7];
  const float* bs   = (const float*)d_in[8];
  const float* Wout = (const float*)d_in[9];
  const float* bout = (const float*)d_in[10];
  const float* Wq   = (const float*)d_in[11];
  const float* Wk   = (const float*)d_in[12];
  const float* Wv   = (const float*)d_in[13];
  const float* Wo   = (const float*)d_in[14];
  const float* Wint = (const float*)d_in[15];
  const float* Wbyp = (const float*)d_in[16];
  const float* bbyp = (const float*)d_in[17];
  const float* c1w  = (const float*)d_in[18];
  const float* c1b  = (const float*)d_in[19];
  const float* c2w  = (const float*)d_in[20];
  const float* c2b  = (const float*)d_in[21];
  const float* Wa1  = (const float*)d_in[22];
  const float* ba1  = (const float*)d_in[23];
  const float* Wa2  = (const float*)d_in[24];
  const float* ba2  = (const float*)d_in[25];

  char* ws = (char*)d_ws;
  const size_t IMGB = (size_t)BZ * CC * HHD * WWD * 4;  // 64 MB
  float* AMPV = (float*)(ws + 0 * IMGB);
  float* PHV  = (float*)(ws + 1 * IMGB);
  float* AMPI = (float*)(ws + 2 * IMGB);
  float* PHI  = (float*)(ws + 3 * IMGB);
  float* TRE  = (float*)(ws + 4 * IMGB);   // FFT tmp re / attention scores / gate matrix
  float* TIM  = (float*)(ws + 5 * IMGB);   // FFT tmp im / bf16 attention probs
  float* FAMP = (float*)(ws + 6 * IMGB);
  float* FPH  = (float*)(ws + 7 * IMGB);
  unsigned short* CATB = (unsigned short*)(ws + 8 * IMGB);  // bf16 [vt|it] tokens, 64 MB
  size_t o = 9 * IMGB;
  auto alloc = [&](size_t bytes) { size_t r = o; o = (o + bytes + 255) & ~(size_t)255; return r; };
  float* POOLED = (float*)(ws + alloc((size_t)BZ * 2 * CC * 4));
  float* INTENT = (float*)(ws + alloc((size_t)BZ * PRI_D * 4));
  float* GAMMA  = (float*)(ws + alloc((size_t)BZ * CC * 4));
  float* BETA   = (float*)(ws + alloc((size_t)BZ * CC * 4));
  float* PRIv   = (float*)(ws + alloc((size_t)BZ * HIDN * 4));
  float* INTv   = (float*)(ws + alloc((size_t)BZ * HIDN * 4));
  float* HBUF   = (float*)(ws + alloc((size_t)BZ * NT * HIDN * 4));
  float* SCORE  = (float*)(ws + alloc((size_t)BZ * NT * 4));
  int*   IDX    = (int*)(ws + alloc((size_t)BZ * KSEL * 4));
  unsigned short* VSB   = (unsigned short*)(ws + alloc((size_t)BZ * KSEL * TOK * 2));
  unsigned short* IRSB  = (unsigned short*)(ws + alloc((size_t)BZ * KSEL * TOK * 2));
  float* QF = (float*)(ws + alloc((size_t)BZ * KSEL * HIDN * 4));
  float* KF = (float*)(ws + alloc((size_t)BZ * KSEL * HIDN * 4));
  float* VF = (float*)(ws + alloc((size_t)BZ * KSEL * HIDN * 4));
  unsigned short* QB  = (unsigned short*)(ws + alloc((size_t)BZ * KSEL * HIDN * 2));
  unsigned short* VB  = (unsigned short*)(ws + alloc((size_t)BZ * KSEL * HIDN * 2));
  unsigned short* KTm = (unsigned short*)(ws + alloc((size_t)BZ * NHEAD * DHD * KSEL * 2));
  float* OF = (float*)(ws + alloc((size_t)BZ * KSEL * HIDN * 4));
  unsigned short* OB = (unsigned short*)(ws + alloc((size_t)BZ * KSEL * HIDN * 2));
  float* FUSED = (float*)(ws + alloc((size_t)BZ * KSEL * TOK * 4));
  unsigned short* WTOKB = (unsigned short*)(ws + alloc((size_t)2048 * HIDN * 2));
  unsigned short* WQB   = (unsigned short*)(ws + alloc((size_t)TOK * HIDN * 2));
  unsigned short* WKB   = (unsigned short*)(ws + alloc((size_t)TOK * HIDN * 2));
  unsigned short* WVB   = (unsigned short*)(ws + alloc((size_t)TOK * HIDN * 2));
  unsigned short* WOB   = (unsigned short*)(ws + alloc((size_t)HIDN * TOK * 2));
  unsigned short* WBYPB = (unsigned short*)(ws + alloc((size_t)2048 * TOK * 2));
  unsigned short* ATTB  = (unsigned short*)TIM;  // bf16 attention probs overlay

  // router
  mean_pool_kernel<<<BZ * 2 * CC, 256, 0, stream>>>(vis, ir, POOLED);
  router_kernel<<<1, 512, 0, stream>>>(POOLED, Wr, br, bank, Wa1, ba1, Wa2, ba2, INTENT, GAMMA, BETA);

  // forward FFT (rows then cols) -> amp/phase per modality
  int fftBlocks = BZ * CC * HHD / 4;  // 16384
  fft256_kernel<<<fftBlocks, 256, 0, stream>>>(vis, nullptr, TRE, TIM, 0, 0, 0, 1.f);
  fft256_kernel<<<fftBlocks, 256, 0, stream>>>(TRE, TIM, AMPV, PHV, 1, 0, 1, 1.f);
  fft256_kernel<<<fftBlocks, 256, 0, stream>>>(ir, nullptr, TRE, TIM, 0, 0, 0, 1.f);
  fft256_kernel<<<fftBlocks, 256, 0, stream>>>(TRE, TIM, AMPI, PHI, 1, 0, 1, 1.f);

  for (int bri = 0; bri < 2; ++bri) {
    const float* srcV = bri ? PHV : AMPV;
    const float* srcI = bri ? PHI : AMPI;
    float* target = bri ? FPH : FAMP;

    long nWtok = 2048L * HIDN;
    cvt_bf16_kernel<<<(unsigned)((nWtok + 255) / 256), 256, 0, stream>>>(Wtok + (size_t)bri * nWtok, WTOKB, nWtok);
    long nWqk = (long)TOK * HIDN;
    cvt_bf16_kernel<<<(unsigned)((nWqk + 255) / 256), 256, 0, stream>>>(Wq + (size_t)bri * nWqk, WQB, nWqk);
    cvt_bf16_kernel<<<(unsigned)((nWqk + 255) / 256), 256, 0, stream>>>(Wk + (size_t)bri * nWqk, WKB, nWqk);
    cvt_bf16_kernel<<<(unsigned)((nWqk + 255) / 256), 256, 0, stream>>>(Wv + (size_t)bri * nWqk, WVB, nWqk);
    long nWo = (long)HIDN * TOK;
    cvt_bf16_kernel<<<(unsigned)((nWo + 255) / 256), 256, 0, stream>>>(Wo + (size_t)bri * nWo, WOB, nWo);
    long nWb = 2048L * TOK;
    cvt_bf16_kernel<<<(unsigned)((nWb + 255) / 256), 256, 0, stream>>>(Wbyp + (size_t)bri * nWb, WBYPB, nWb);
    intent_proj_kernel<<<1, 512, 0, stream>>>(INTENT, Wpri + (size_t)bri * PRI_D * HIDN,
                                              Wint + (size_t)bri * PRI_D * HIDN, PRIv, INTv);

    size_t npack = (size_t)BZ * NT * 2048;
    pack_tokens_kernel<<<(unsigned)(npack / 256), 256, 0, stream>>>(srcV, srcI, CATB);

    // H = [vt|it] @ Wtok : 16384 x 128 x 2048
    dim3 g1(2, 256, 1);
    gemm_bf16_fast_kernel<<<g1, 128, 0, stream>>>(CATB, WTOKB, HBUF, 2048, 2048, 128, 128,
                                                  0, 0, 0, 0, 0, 0, 1, 1.f, 0);
    score_kernel<<<BZ * NT, 128, 0, stream>>>(HBUF, PRIv, Wco + (size_t)bri * 2 * HIDN,
                                              bs + (size_t)bri * HIDN, Wout + (size_t)bri * HIDN,
                                              bout + bri, SCORE);
    topk_kernel<<<BZ, 1024, 0, stream>>>(SCORE, IDX);

    size_t ng = (size_t)BZ * KSEL * TOK;
    gather_kernel<<<(unsigned)(ng / 256), 256, 0, stream>>>(CATB, srcV, IDX, VSB, IRSB, FUSED);

    // Q/K/V projections: 4096 x 128 x 1024
    dim3 gq(2, 64, 1);
    gemm_bf16_fast_kernel<<<gq, 128, 0, stream>>>(VSB, WQB, QF, 1024, 1024, 128, 128,
                                                  0, 0, 0, 0, 0, 0, 1, 1.f, 0);
    gemm_bf16_fast_kernel<<<gq, 128, 0, stream>>>(IRSB, WKB, KF, 1024, 1024, 128, 128,
                                                  0, 0, 0, 0, 0, 0, 1, 1.f, 0);
    gemm_bf16_fast_kernel<<<gq, 128, 0, stream>>>(IRSB, WVB, VF, 1024, 1024, 128, 128,
                                                  0, 0, 0, 0, 0, 0, 1, 1.f, 0);
    long nqb = 4096L * HIDN;
    cvt_bf16_kernel<<<(unsigned)((nqb + 255) / 256), 256, 0, stream>>>(QF, QB, nqb);
    cvt_bf16_kernel<<<(unsigned)((nqb + 255) / 256), 256, 0, stream>>>(VF, VB, nqb);
    kT_kernel<<<(unsigned)(((size_t)BZ * NHEAD * DHD * KSEL) / 256), 256, 0, stream>>>(KF, KTm);

    // scores[b,h] = Q_bh @ K_bh^T / sqrt(dh) : 1024 x 1024 x 32, batched over 16
    dim3 gs(16, 16, BZ * NHEAD);
    gemm_bf16_fast_kernel<<<gs, 128, 0, stream>>>(QB, KTm, TRE, 32, 128, 1024, 1024,
        (long long)KSEL * HIDN, (long long)DHD,
        (long long)NHEAD * DHD * KSEL, (long long)DHD * KSEL,
        (long long)NHEAD * KSEL * KSEL, (long long)KSEL * KSEL,
        NHEAD, 0.17677669529663687f, 0);
    softmax_kernel<<<BZ * NHEAD * KSEL, 256, 0, stream>>>(TRE, ATTB);

    // O[b,h] = attn @ V_bh : 1024 x 32 x 1024, batched over 16 (dedicated N=32 kernel)
    dim3 ga(1, 16, BZ * NHEAD);
    gemm_bf16_n32_kernel<<<ga, 128, 0, stream>>>(ATTB, VB, OF, 1024, 1024, 128, 128,
        (long long)NHEAD * KSEL * KSEL, (long long)KSEL * KSEL,
        (long long)KSEL * HIDN, (long long)DHD,
        (long long)KSEL * HIDN, (long long)DHD,
        NHEAD, 1.f);
    o_post_kernel<<<(unsigned)(((size_t)BZ * KSEL * HIDN) / 256), 256, 0, stream>>>(OF, INTv, OB);

    // FUSED(=vs) += O @ Wo : 4096 x 1024 x 128
    dim3 go(16, 64, 1);
    gemm_bf16_fast_kernel<<<go, 128, 0, stream>>>(OB, WOB, FUSED, 128, 128, 1024, 1024,
                                                  0, 0, 0, 0, 0, 0, 1, 1.f, 1);

    // bypass gate: 16384 x 1024 x 2048 (largest GEMM) -> TRE
    dim3 gb(16, 256, 1);
    gemm_bf16_fast_kernel<<<gb, 128, 0, stream>>>(CATB, WBYPB, TRE, 2048, 2048, 1024, 1024,
                                                  0, 0, 0, 0, 0, 0, 1, 1.f, 0);
    size_t nbl = (size_t)BZ * NT * TOK;
    blend_kernel<<<(unsigned)(nbl / 256), 256, 0, stream>>>(TRE, bbyp + (size_t)bri * TOK, srcV, srcI, target);
    scatter_kernel<<<(unsigned)(ng / 256), 256, 0, stream>>>(FUSED, IDX, target);
  }

  // spectrum rebuild + inverse FFT (conjugate trick) -> spatial
  size_t npix = (size_t)BZ * CC * HHD * WWD;
  spec_kernel<<<(unsigned)(npix / 256), 256, 0, stream>>>(FAMP, FPH, AMPV, PHV);
  fft256_kernel<<<fftBlocks, 256, 0, stream>>>(AMPV, PHV, TRE, TIM, 0, 1, 0, 1.f);
  fft256_kernel<<<fftBlocks, 256, 0, stream>>>(TRE, TIM, AMPI, nullptr, 1, 0, 2, 1.f / 65536.f);

  // conv1 (relu) -> PHI ; conv2 + affine + residual -> d_out
  dim3 gc(WWD / 16, HHD / 16, BZ * CC);
  conv3_kernel<<<gc, 256, 0, stream>>>(AMPI, c1w, c1b, PHI, 0, nullptr, nullptr, nullptr, nullptr);
  conv3_kernel<<<gc, 256, 0, stream>>>(PHI, c2w, c2b, (float*)d_out, 1, GAMMA, BETA, vis, ir);
}
